// PicassoNetII_54640573939788
// MI455X (gfx1250) — matrix operand order repaired and checked by reading, not compiled, not run on hardware
//
#include <hip/hip_runtime.h>

typedef __attribute__((ext_vector_type(16))) __bf16 v16bf;
typedef __attribute__((ext_vector_type(8)))  __bf16 v8bf;
typedef __attribute__((ext_vector_type(8)))  float  v8f;

#define EPSF 1e-8f

// ---------------------------------------------------------------------------
// Kernel 1: per-face geometry + SH coefficients + first MLP (h = relu(geo@W_geo+b))
// Writes h (Nf x 32) and coeff (Nf x 16) in bf16 for the WMMA GEMM.
// ---------------------------------------------------------------------------
__global__ void face_prep_kernel(const float* __restrict__ vertex,
                                 const int*   __restrict__ face,
                                 const float* __restrict__ normals,
                                 const float* __restrict__ W_geo,   // 9 x 32
                                 const float* __restrict__ b_geo,   // 32
                                 __bf16* __restrict__ h_out,        // Nf x 32
                                 __bf16* __restrict__ c_out,        // Nf x 16
                                 int Nf)
{
    int f = blockIdx.x * blockDim.x + threadIdx.x;
    if (f >= Nf) return;

    // normalized normal
    float nx = normals[f*3+0], ny = normals[f*3+1], nz = normals[f*3+2];
    float inv = 1.0f / (sqrtf(nx*nx + ny*ny + nz*nz) + EPSF);
    nx *= inv; ny *= inv; nz *= inv;

    // triangle geometry
    int i1 = face[f*3+0], i2 = face[f*3+1], i3 = face[f*3+2];
    float v1x = vertex[i1*3+0], v1y = vertex[i1*3+1], v1z = vertex[i1*3+2];
    float v2x = vertex[i2*3+0], v2y = vertex[i2*3+1], v2z = vertex[i2*3+2];
    float v3x = vertex[i3*3+0], v3y = vertex[i3*3+1], v3z = vertex[i3*3+2];

    float d12x = v2x-v1x, d12y = v2y-v1y, d12z = v2z-v1z;
    float d23x = v3x-v2x, d23y = v3y-v2y, d23z = v3z-v2z;
    float d31x = v1x-v3x, d31y = v1y-v3y, d31z = v1z-v3z;

    float L12 = sqrtf(d12x*d12x + d12y*d12y + d12z*d12z) + EPSF;
    float L23 = sqrtf(d23x*d23x + d23y*d23y + d23z*d23z) + EPSF;
    float L31 = sqrtf(d31x*d31x + d31y*d31y + d31z*d31z) + EPSF;

    float T1 = -(d12x*d31x + d12y*d31y + d12z*d31z) / (L12*L31);
    float T2 = -(d12x*d23x + d12y*d23y + d12z*d23z) / (L12*L23);
    float T3 = -(d23x*d31x + d23y*d31y + d23z*d31z) / (L23*L31);

    float geo[9] = {L12, L23, L31, T1, T2, T3, nx, ny, nz};

    // h = relu(geo @ W_geo + b_geo); W_geo uniform -> scalar-cached loads
    #pragma unroll
    for (int o = 0; o < 32; ++o) {
        float acc = b_geo[o];
        #pragma unroll
        for (int i = 0; i < 9; ++i) acc += geo[i] * W_geo[i*32 + o];
        h_out[f*32 + o] = (__bf16)fmaxf(acc, 0.0f);
    }

    // spherical harmonic coefficients (16)
    float x = nx, y = ny, z = nz;
    float x2 = x*x, y2 = y*y, z2 = z*z;
    float c[16];
    c[0]  = 0.282095f;
    c[1]  = 0.488603f * y;
    c[2]  = 0.488603f * z;
    c[3]  = 0.488603f * x;
    c[4]  = 1.092548f * x * y;
    c[5]  = 1.092548f * y * z;
    c[6]  = 0.315392f * (3.0f*z2 - 1.0f);
    c[7]  = 1.092548f * x * z;
    c[8]  = 0.546274f * (x2 - y2);
    c[9]  = 0.590044f * y * (3.0f*x2 - y2);
    c[10] = 2.890611f * x * y * z;
    c[11] = 0.457046f * y * (5.0f*z2 - 1.0f);
    c[12] = 0.373176f * z * (5.0f*z2 - 3.0f);
    c[13] = 0.457046f * x * (5.0f*z2 - 1.0f);
    c[14] = 1.445306f * z * (x2 - y2);
    c[15] = 0.590044f * x * (x2 - 3.0f*y2);
    #pragma unroll
    for (int k = 0; k < 16; ++k) c_out[f*16 + k] = (__bf16)c[k];
}

// ---------------------------------------------------------------------------
// Kernel 2: fused GEMM  contrib = u @ W_flat + b_sph   (M=Nf, K=512, N=32)
// u[f, k*32+i] = coeff[f,k]*h[f,i] is formed in registers per K-chunk as a
// bf16 vector x scalar product (v_pk_mul_bf16 + op_sel: splat is free).
// W_sph (64KB f32) is pre-swizzled into LDS as bf16 B-fragments (padded to
// 48B/lane to break the 4-way bank conflict of a 32B stride).
// Each wave processes TWO 16-face tiles per iteration: per K-chunk the same
// pair of B fragments feeds 4 independent WMMAs (2 face-tiles x 2 N-tiles).
// A sched_barrier per K-chunk (mask: DS-read|SALU may cross) stops the
// scheduler from hoisting all 32 chunk-invariant A-builds/B-loads at once,
// which previously either blew past 256 VGPRs (round 2) or spilled to
// scratch under a hard waves_per_eu cap (round 3). B-load prefetch across
// chunks remains allowed.
// ---------------------------------------------------------------------------
#define LANE_PAD   24                    // bf16 elems per lane block (48B)
#define CHUNK_STR  (32 * LANE_PAD)       // 768 elems per chunk
#define TILE_STR   (16 * CHUNK_STR)      // 12288 elems per N-tile
#define LDSB_ELEMS (2 * TILE_STR)        // 24576 elems = 48KB

__global__ __launch_bounds__(256)
void sph_gemm_kernel(
    const __bf16* __restrict__ h_in,     // Nf x 32
    const __bf16* __restrict__ c_in,     // Nf x 16
    const float*  __restrict__ W_sph,    // 16 x 32 x 32
    const float*  __restrict__ b_sph,    // 32
    float* __restrict__ contrib,         // Nf x 32
    int Nf, int nPairs)
{
    __shared__ __bf16 ldsB[LDSB_ELEMS];

    int tid = threadIdx.x;
    // preload + swizzle B: logical [tile][ck][lane][e] -> padded lane blocks
    for (int j = tid; j < 16384; j += 256) {
        int tile = j >> 13;
        int r    = j & 8191;
        int ck   = r >> 9;
        int r2   = r & 511;
        int lane = r2 >> 4;
        int e    = r2 & 15;
        int i    = ((e >> 3) << 4) + ((lane >> 4) << 3) + (e & 7);  // B row within chunk: K = 16*(e/8) + 8*hs + e%8
        int o    = (tile << 4) + (lane & 15);     // output column
        ldsB[tile*TILE_STR + ck*CHUNK_STR + lane*LANE_PAD + e] =
            (__bf16)W_sph[(ck*32 + i)*32 + o];
    }
    __syncthreads();

    int wid  = tid >> 5;
    int lane = tid & 31;
    int M    = lane & 15;
    int hs   = lane >> 4;
    int stride = gridDim.x * 8;

    for (int p = blockIdx.x * 8 + wid; p < nPairs; p += stride) {
        int fbase0 = p * 32;          // face tile 0
        int fbase1 = p * 32 + 16;     // face tile 1
        int fA0 = fbase0 + M;
        int fA1 = fbase1 + M;
        int fc0 = fA0 < Nf ? fA0 : Nf - 1;
        int fc1 = fA1 < Nf ? fA1 : Nf - 1;

        // A-side per-lane data, kept in packed bf16 (8 VGPRs per v16bf)
        union { v16bf v; v8bf h[2]; } h0u, h1u, c0u, c1u;
        h0u.h[0] = *(const v8bf*)(h_in + (size_t)fc0*32 + hs*8);
        h0u.h[1] = *(const v8bf*)(h_in + (size_t)fc0*32 + 16 + hs*8);
        h1u.h[0] = *(const v8bf*)(h_in + (size_t)fc1*32 + hs*8);
        h1u.h[1] = *(const v8bf*)(h_in + (size_t)fc1*32 + 16 + hs*8);
        c0u.h[0] = *(const v8bf*)(c_in + (size_t)fc0*16);
        c0u.h[1] = *(const v8bf*)(c_in + (size_t)fc0*16 + 8);
        c1u.h[0] = *(const v8bf*)(c_in + (size_t)fc1*16);
        c1u.h[1] = *(const v8bf*)(c_in + (size_t)fc1*16 + 8);

        // tail mask: zero the coeffs of out-of-range faces -> zero contribution
        v16bf zero = {};
        v16bf c0v = (fA0 < Nf) ? c0u.v : zero;
        v16bf c1v = (fA1 < Nf) ? c1u.v : zero;
        v16bf h0v = h0u.v;
        v16bf h1v = h1u.v;

        // accumulators pre-loaded with bias: C-layout gives every element of a
        // lane's accumulator the same N column, so bias is a per-lane splat.
        float bs0 = b_sph[lane & 15];
        float bs1 = b_sph[16 + (lane & 15)];
        v8f acc00, acc01, acc10, acc11;
        #pragma unroll
        for (int r = 0; r < 8; ++r) {
            acc00[r] = bs0; acc01[r] = bs1;
            acc10[r] = bs0; acc11[r] = bs1;
        }

        const __bf16* bbase = &ldsB[lane * LANE_PAD];

        #pragma unroll
        for (int ck = 0; ck < 16; ++ck) {
            // fence: keep this chunk's VALU/WMMA in place; DS reads (0x100)
            // and SALU (0x4) may still move across for B prefetch.
            __builtin_amdgcn_sched_barrier(0x104);

            // shared B fragments for this chunk (both N-tiles)
            union { v16bf v; v8bf h[2]; } b0, b1;
            const v8bf* bp0 = (const v8bf*)(bbase + ck*CHUNK_STR);
            const v8bf* bp1 = (const v8bf*)(bbase + TILE_STR + ck*CHUNK_STR);
            b0.h[0] = bp0[0]; b0.h[1] = bp0[1];
            b1.h[0] = bp1[0]; b1.h[1] = bp1[1];

            // A fragments: bf16 rank-1 scale (vector x scalar, op_sel splat)
            __bf16 s0 = c0v[ck];
            __bf16 s1 = c1v[ck];
            v16bf a0 = h0v * s0;
            v16bf a1 = h1v * s1;

            acc00 = __builtin_amdgcn_wmma_f32_16x16x32_bf16(
                        false, a0, false, b0.v, (short)0, acc00, false, false);
            acc10 = __builtin_amdgcn_wmma_f32_16x16x32_bf16(
                        false, a1, false, b0.v, (short)0, acc10, false, false);
            acc01 = __builtin_amdgcn_wmma_f32_16x16x32_bf16(
                        false, a0, false, b1.v, (short)0, acc01, false, false);
            acc11 = __builtin_amdgcn_wmma_f32_16x16x32_bf16(
                        false, a1, false, b1.v, (short)0, acc11, false, false);
        }

        #pragma unroll
        for (int r = 0; r < 8; ++r) {
            int f0 = fbase0 + r + 8*hs;   // C layout: VGPR r -> M = r + 8*hs
            int f1 = fbase1 + r + 8*hs;
            if (f0 < Nf) {
                contrib[(size_t)f0*32 + (lane & 15)]      = acc00[r];
                contrib[(size_t)f0*32 + 16 + (lane & 15)] = acc01[r];
            }
            if (f1 < Nf) {
                contrib[(size_t)f1*32 + (lane & 15)]      = acc10[r];
                contrib[(size_t)f1*32 + 16 + (lane & 15)] = acc11[r];
            }
        }
    }
}

// ---------------------------------------------------------------------------
// Kernel 3: scatter contrib to vertices (segment_sum via f32 atomics) + counts
// ---------------------------------------------------------------------------
__global__ void scatter_kernel(const float* __restrict__ contrib,
                               const int*   __restrict__ face,
                               float* __restrict__ vfeat,   // Nv x 32
                               float* __restrict__ vcnt,    // Nv
                               int Nf)
{
    int t = blockIdx.x * blockDim.x + threadIdx.x;
    if (t >= Nf * 32) return;
    int f = t >> 5, o = t & 31;
    float val = contrib[t];
    int v0 = face[f*3+0], v1 = face[f*3+1], v2 = face[f*3+2];
    atomicAdd(&vfeat[(size_t)v0*32 + o], val);
    atomicAdd(&vfeat[(size_t)v1*32 + o], val);
    atomicAdd(&vfeat[(size_t)v2*32 + o], val);
    if (o == 0) {
        atomicAdd(&vcnt[v0], 1.0f);
        atomicAdd(&vcnt[v1], 1.0f);
        atomicAdd(&vcnt[v2], 1.0f);
    }
}

// ---------------------------------------------------------------------------
// Kernel 4: per-vertex mean + relu + final projection (32 -> 20)
// ---------------------------------------------------------------------------
__global__ void pred_kernel(const float* __restrict__ vfeat,
                            const float* __restrict__ vcnt,
                            const float* __restrict__ W_pred,  // 32 x 20
                            const float* __restrict__ b_pred,  // 20
                            float* __restrict__ out,           // Nv x 20
                            int Nv)
{
    int v = blockIdx.x * blockDim.x + threadIdx.x;
    if (v >= Nv) return;
    float inv = 1.0f / (vcnt[v] + EPSF);
    float a[32];
    #pragma unroll
    for (int o = 0; o < 32; ++o) a[o] = fmaxf(vfeat[(size_t)v*32 + o] * inv, 0.0f);
    #pragma unroll
    for (int nc = 0; nc < 20; ++nc) {
        float acc = b_pred[nc];
        #pragma unroll
        for (int o = 0; o < 32; ++o) acc += a[o] * W_pred[o*20 + nc];
        out[(size_t)v*20 + nc] = acc;
    }
}

// ---------------------------------------------------------------------------
__global__ void zero_kernel(float* __restrict__ p, int n)
{
    int t = blockIdx.x * blockDim.x + threadIdx.x;
    if (t < n) p[t] = 0.0f;
}

// ---------------------------------------------------------------------------
extern "C" void kernel_launch(void* const* d_in, const int* in_sizes, int n_in,
                              void* d_out, int out_size, void* d_ws, size_t ws_size,
                              hipStream_t stream)
{
    const float* vertex  = (const float*)d_in[0];
    const int*   face    = (const int*)  d_in[1];
    const float* normals = (const float*)d_in[2];
    const float* W_geo   = (const float*)d_in[3];
    const float* b_geo   = (const float*)d_in[4];
    const float* W_sph   = (const float*)d_in[5];
    const float* b_sph   = (const float*)d_in[6];
    const float* W_pred  = (const float*)d_in[7];
    const float* b_pred  = (const float*)d_in[8];
    float* out = (float*)d_out;

    int Nv = in_sizes[0] / 3;
    int Nf = in_sizes[1] / 3;
    int nTiles = (Nf + 15) / 16;
    int nPairs = (nTiles + 1) / 2;

    // workspace carve-out (all 256B-aligned)
    char* ws = (char*)d_ws;
    size_t off = 0;
    auto take = [&](size_t bytes) -> char* {
        char* p = ws + off;
        off = (off + bytes + 255) & ~(size_t)255;
        return p;
    };
    __bf16* h_bf    = (__bf16*)take((size_t)Nf * 32 * sizeof(__bf16));
    __bf16* c_bf    = (__bf16*)take((size_t)Nf * 16 * sizeof(__bf16));
    float*  contrib = (float*) take((size_t)Nf * 32 * sizeof(float));
    float*  vfeat   = (float*) take((size_t)Nv * 32 * sizeof(float));
    float*  vcnt    = (float*) take((size_t)Nv * sizeof(float));

    // 0) zero accumulators (must re-zero every call: atomics accumulate)
    {
        int n = Nv * 32;
        zero_kernel<<<(n + 255) / 256, 256, 0, stream>>>(vfeat, n);
        zero_kernel<<<(Nv + 255) / 256, 256, 0, stream>>>(vcnt, Nv);
    }

    // 1) per-face prep
    face_prep_kernel<<<(Nf + 255) / 256, 256, 0, stream>>>(
        vertex, face, normals, W_geo, b_geo, h_bf, c_bf, Nf);

    // 2) fused WMMA GEMM (M=Nf, K=512, N=32), bf16 inputs, f32 accumulate
    {
        int blocks = (nPairs + 7) / 8;
        if (blocks > 384) blocks = 384;   // amortize LDS B preload over pairs
        sph_gemm_kernel<<<blocks, 256, 0, stream>>>(
            h_bf, c_bf, W_sph, b_sph, contrib, Nf, nPairs);
    }

    // 3) scatter to vertices
    {
        int n = Nf * 32;
        scatter_kernel<<<(n + 255) / 256, 256, 0, stream>>>(
            contrib, face, vfeat, vcnt, Nf);
    }

    // 4) mean + relu + final projection
    pred_kernel<<<(Nv + 255) / 256, 256, 0, stream>>>(
        vfeat, vcnt, W_pred, b_pred, out, Nv);
}
